// Qwen3Attention_14044543058103
// MI455X (gfx1250) — matrix operand order repaired and checked by reading, not compiled, not run on hardware
//
#include <hip/hip_runtime.h>
#include <hip/hip_bf16.h>

typedef __attribute__((ext_vector_type(16))) __bf16        v16bf;
typedef __attribute__((ext_vector_type(8)))  float         v8f;
typedef __attribute__((ext_vector_type(8)))  unsigned int  v8u;
typedef __attribute__((ext_vector_type(4)))  unsigned int  v4u;
typedef __attribute__((ext_vector_type(8)))  unsigned short ush8;

// ---------------- problem constants ----------------
constexpr int kB   = 2;
constexpr int kT   = 2048;
constexpr int kD   = 1024;
constexpr int kHD  = 128;
constexpr int kNH  = 16;
constexpr int kNKV = 8;
constexpr float kScale = 0.08838834764831845f;  // 1/sqrt(128)
constexpr float kEps   = 1e-6f;

// ---------------- Tensor Data Mover: 2D tile global -> LDS ----------------
// Builds the D# descriptor (ISA 08_async_tensor §8.3/8.4) and issues
// tensor_load_to_lds. All inputs must be wave-uniform. data_size = 2 bytes.
__device__ __forceinline__ void tdm_load_2d_bf16(const void* gptr, void* lptr,
                                                 unsigned tile_w, unsigned tile_h,
                                                 unsigned stride_elems,
                                                 unsigned tensor_w, unsigned tensor_h) {
  unsigned long long ga = (unsigned long long)gptr;
  unsigned lds = (unsigned)(unsigned long long)lptr;  // addr[31:0] = LDS byte offset
  v4u g0;
  g0[0] = 1u;                                               // count=1 (user mode)
  g0[1] = lds;                                              // lds_addr
  g0[2] = (unsigned)ga;                                     // global_addr[31:0]
  g0[3] = (unsigned)((ga >> 32) & 0x1FFFFFFull) | (2u << 30); // addr[56:32] | type=2
  v8u g1;
  g1[0] = 1u << 16;                                         // data_size=1 (2 bytes)
  g1[1] = (tensor_w & 0xFFFFu) << 16;                       // tensor_dim0[15:0]
  g1[2] = (tensor_w >> 16) | ((tensor_h & 0xFFFFu) << 16);  // dim0 hi | dim1 lo
  g1[3] = (tensor_h >> 16) | (tile_w << 16);                // dim1 hi | tile_dim0
  g1[4] = tile_h;                                           // tile_dim1 (tile_dim2=0)
  g1[5] = stride_elems;                                     // tensor_dim0_stride[31:0]
  g1[6] = 0;
  g1[7] = 0;
  asm volatile("tensor_load_to_lds %0, %1" :: "s"(g0), "s"(g1) : "memory");
}

// ---------------- f32 -> bf16 convert ----------------
__global__ __launch_bounds__(256)
void cvt_f32_bf16(const float* __restrict__ in, __bf16* __restrict__ out, int n) {
  int i = blockIdx.x * 256 + threadIdx.x;
  if (i < n) out[i] = (__bf16)in[i];
}

// ---------------- WMMA GEMM: C[M,N] = A[M,K] * B[K,N] ----------------
// A,B bf16 row-major. Block tile 128x128x32; 256 threads = 8 waves,
// waves arranged 4(M) x 2(N); each wave owns 2x4 16x16 WMMA tiles.
// Double-buffered LDS: A tiles DMA'd by the Tensor Data Mover, B tiles staged
// transposed through registers — the (i+1) tile is in flight during tile i's
// WMMA burst, so DMA/global latency hides under compute.
template <typename OutT>
__global__ __launch_bounds__(256)
void gemm_bf16_wmma(const __bf16* __restrict__ A, const __bf16* __restrict__ B,
                    OutT* __restrict__ C, int M, int N, int K) {
  constexpr int BM = 128, BN = 128, BK = 32;
  __shared__ __bf16 As[2][BM][BK];   // 16 KB, filled by TDM (ping-pong)
  __shared__ __bf16 Bt[2][BN][BK];   // 16 KB, transposed B tiles (ping-pong)

  const int m0   = blockIdx.x * BM;
  const int n0   = blockIdx.y * BN;
  const int tid  = threadIdx.x;
  const int wave = tid >> 5;
  const int lane = tid & 31;
  const int wm   = wave & 3;   // 0..3
  const int wn   = wave >> 2;  // 0..1
  const int half = lane >> 4;
  const int l15  = lane & 15;
  const int kk   = tid & 31;          // B staging row (k)
  const int nb   = (tid >> 5) * 16;   // B staging col group (n)

  v8f acc[2][4] = {};
  const int nIter = K / BK;

  // ---- prologue: stage tile 0 into buffer 0
  if (wave == 0)
    tdm_load_2d_bf16(A + (size_t)m0 * K, &As[0][0][0], BK, BM, K, K, M);
  {
    const __bf16* src = B + (size_t)kk * N + (n0 + nb);
    ush8 b0 = *(const ush8*)(src);
    ush8 b1 = *(const ush8*)(src + 8);
    const __bf16* e0 = (const __bf16*)&b0;
    const __bf16* e1 = (const __bf16*)&b1;
#pragma unroll
    for (int e = 0; e < 8; ++e) Bt[0][nb + e][kk] = e0[e];
#pragma unroll
    for (int e = 0; e < 8; ++e) Bt[0][nb + 8 + e][kk] = e1[e];
  }

  for (int i = 0; i < nIter; ++i) {
    const int cur = i & 1;
    const int k0  = i * BK;
    if (wave == 0) __builtin_amdgcn_s_wait_tensorcnt(0);
    __syncthreads();  // As[cur]/Bt[cur] published; prior reads of [cur^1] done

    // ---- kick off tile i+1 into the alternate buffer (overlaps compute)
    const bool havenext = (i + 1 < nIter);
    ush8 nb0, nb1;
    if (havenext) {
      if (wave == 0)
        tdm_load_2d_bf16(A + (size_t)m0 * K + (k0 + BK), &As[cur ^ 1][0][0],
                         BK, BM, K, K, M);
      const __bf16* src = B + (size_t)(k0 + BK + kk) * N + (n0 + nb);
      nb0 = *(const ush8*)(src);
      nb1 = *(const ush8*)(src + 8);
    }

    // ---- load ALL fragments first, then issue 8 WMMAs back-to-back
    v16bf afrag[2], bfrag[4];
#pragma unroll
    for (int mt = 0; mt < 2; ++mt) {
      const int r = wm * 32 + mt * 16 + l15;
      v8u au;
#pragma unroll
      for (int p = 0; p < 8; ++p) {
        const int kbp = 2 * p + 16 * half;  // A-frag K pattern
        au[p] = *(const unsigned int*)&As[cur][r][kbp];
      }
      afrag[mt] = __builtin_bit_cast(v16bf, au);
    }
#pragma unroll
    for (int nt = 0; nt < 4; ++nt) {
      const int n = wn * 64 + nt * 16 + l15;
      v8u bu;
#pragma unroll
      for (int p = 0; p < 8; ++p)
        bu[p] = *(const unsigned int*)&Bt[cur][n][2 * p + 16 * half];  // B-frag pattern
      bfrag[nt] = __builtin_bit_cast(v16bf, bu);
    }
#pragma unroll
    for (int mt = 0; mt < 2; ++mt)
#pragma unroll
      for (int nt = 0; nt < 4; ++nt)
        acc[mt][nt] = __builtin_amdgcn_wmma_f32_16x16x32_bf16(
            false, afrag[mt], false, bfrag[nt], (short)0, acc[mt][nt], false, false);

    // ---- scatter the staged B registers for tile i+1 (loads are done by now)
    if (havenext) {
      const __bf16* e0 = (const __bf16*)&nb0;
      const __bf16* e1 = (const __bf16*)&nb1;
#pragma unroll
      for (int e = 0; e < 8; ++e) Bt[cur ^ 1][nb + e][kk] = e0[e];
#pragma unroll
      for (int e = 0; e < 8; ++e) Bt[cur ^ 1][nb + 8 + e][kk] = e1[e];
    }
    __syncthreads();  // reads of [cur] done before next iter writes it
  }

  // ---- store: C layout row = j + 8*half, col = lane&15
#pragma unroll
  for (int mt = 0; mt < 2; ++mt)
#pragma unroll
    for (int nt = 0; nt < 4; ++nt)
#pragma unroll
      for (int j = 0; j < 8; ++j) {
        const int row = m0 + wm * 32 + mt * 16 + j + 8 * half;
        const int col = n0 + wn * 64 + nt * 16 + l15;
        C[(size_t)row * N + col] = (OutT)acc[mt][nt][j];
      }
}

// ---------------- fused RMSNorm + RoPE ----------------
// in : bf16 [B, T, H, HD]   out : bf16 [B, H, T, HD]
__global__ __launch_bounds__(128)
void norm_rope_kernel(const __bf16* __restrict__ in, const float* __restrict__ w,
                      __bf16* __restrict__ out, int H) {
  const int idx = blockIdx.x;
  const int t = idx & (kT - 1);
  const int h = (idx >> 11) % H;
  const int b = idx / (kT * H);
  const int d = threadIdx.x;

  __shared__ float row[kHD];
  __shared__ float red[4];

  const float x = (float)in[(((size_t)b * kT + t) * H + h) * kHD + d];
  row[d] = x;
  float ss = x * x;
#pragma unroll
  for (int o = 16; o > 0; o >>= 1) ss += __shfl_xor(ss, o, 32);
  if ((d & 31) == 0) red[d >> 5] = ss;
  __syncthreads();
  const float total = red[0] + red[1] + red[2] + red[3];
  const float rms = sqrtf(total * (1.0f / kHD));
  const float inv = 1.0f / (rms + kEps);

  const int i = d >> 1;
  const float xr = row[2 * i]     * inv * w[2 * i];
  const float xi = row[2 * i + 1] * inv * w[2 * i + 1];
  // theta_i = ROPE_THETA^(-2i/HD); freq = t * theta_i
  const float freq = (float)t * __expf(-(2.0f * (float)i / (float)kHD) * 13.815510557964274f);
  float s, c;
  __sincosf(freq, &s, &c);
  const float val = (d & 1) ? (xr * s + xi * c) : (xr * c - xi * s);
  out[((((size_t)b * H + h) * kT + t) * kHD) + d] = (__bf16)val;
}

// ---------------- V permute: [B,T,KV,HD] -> [B,KV,T,HD] ----------------
__global__ __launch_bounds__(256)
void vperm_kernel(const __bf16* __restrict__ in, __bf16* __restrict__ out) {
  const int i = blockIdx.x * 256 + threadIdx.x;  // total 2*2048*8*128 = 1<<22
  const int d  = i & 127;
  const int hh = (i >> 7) & 7;
  const int t  = (i >> 10) & 2047;
  const int b  = i >> 21;
  out[((((size_t)b * kNKV + hh) * kT + t) << 7) + d] = in[i];
}

// ---------------- flash attention (causal, GQA 2:1) ----------------
// q: [B,NH,T,HD]  k,v: [B,NKV,T,HD]  y: [B,T,NH*HD]   all bf16
// K/V tiles DMA'd by the Tensor Data Mover (wave0 -> K, wave1 -> V) with
// ping-pong LDS buffers: the next KV block's DMA is in flight during the
// current block's WMMA/softmax work.
__global__ __launch_bounds__(128)
void flash_attn_kernel(const __bf16* __restrict__ q, const __bf16* __restrict__ k,
                       const __bf16* __restrict__ v, __bf16* __restrict__ y) {
  __shared__ __bf16 Ks[2][64][kHD];    // 32 KB
  __shared__ __bf16 Vs[2][64][kHD];    // 32 KB
  __shared__ __bf16 Plds[4][16][64];   // 8 KB, per-wave P tiles

  const int wid  = threadIdx.x >> 5;
  const int lane = threadIdx.x & 31;
  const int half = lane >> 4;
  const int l15  = lane & 15;

  const int qb = blockIdx.x & 31;          // 32 query blocks of 64
  const int h  = (blockIdx.x >> 5) & 15;
  const int b  = blockIdx.x >> 9;
  const int q0 = qb * 64;

  const __bf16* qp = q + ((size_t)(b * kNH + h) * kT) * kHD;
  const __bf16* kp = k + ((size_t)(b * kNKV + (h >> 1)) * kT) * kHD;
  const __bf16* vp = v + ((size_t)(b * kNKV + (h >> 1)) * kT) * kHD;

  // ---- kick off block 0 DMA before doing anything else
  if (wid == 0) tdm_load_2d_bf16(kp, &Ks[0][0][0], kHD, 64, kHD, kHD, kT);
  if (wid == 1) tdm_load_2d_bf16(vp, &Vs[0][0][0], kHD, 64, kHD, kHD, kT);

  // Q fragments for this wave's 16 rows (A-layout, loaded once; overlaps DMA)
  v16bf qfrag[4];
  {
    const __bf16* qrow = qp + (size_t)(q0 + wid * 16 + l15) * kHD;
#pragma unroll
    for (int ks = 0; ks < 4; ++ks) {
      v8u u;
#pragma unroll
      for (int p = 0; p < 8; ++p) {
        const int kbp = ks * 32 + 2 * p + 16 * half;
        u[p] = *(const unsigned int*)(qrow + kbp);
      }
      qfrag[ks] = __builtin_bit_cast(v16bf, u);
    }
  }

  v8f   Oacc[8] = {};
  float rowmax[8], rowsum[8];
#pragma unroll
  for (int j = 0; j < 8; ++j) { rowmax[j] = -3.0e38f; rowsum[j] = 0.0f; }

  for (int kb = 0; kb <= qb; ++kb) {
    const int cur = kb & 1;
    if (wid < 2) __builtin_amdgcn_s_wait_tensorcnt(0);
    __syncthreads();  // Ks/Vs[cur] published; prior reads of [cur^1] done

    // ---- next block's DMA in flight during this block's compute
    if (kb + 1 <= qb) {
      const int nxt = (size_t)(kb + 1) * 64;
      if (wid == 0)
        tdm_load_2d_bf16(kp + (size_t)nxt * kHD, &Ks[cur ^ 1][0][0], kHD, 64, kHD, kHD, kT);
      if (wid == 1)
        tdm_load_2d_bf16(vp + (size_t)nxt * kHD, &Vs[cur ^ 1][0][0], kHD, 64, kHD, kHD, kT);
    }
    const int kv0 = kb * 64;

    // ---- S = Q * K^T  (16 rows x 64 keys per wave)
    v8f S[4] = {};
#pragma unroll
    for (int ks = 0; ks < 4; ++ks) {
      v16bf kf[4];
#pragma unroll
      for (int nt = 0; nt < 4; ++nt) {
        v8u u;
#pragma unroll
        for (int p = 0; p < 8; ++p)
          u[p] = *(const unsigned int*)&Ks[cur][nt * 16 + l15][ks * 32 + 2 * p + 16 * half];
        kf[nt] = __builtin_bit_cast(v16bf, u);
      }
#pragma unroll
      for (int nt = 0; nt < 4; ++nt)
        S[nt] = __builtin_amdgcn_wmma_f32_16x16x32_bf16(
            false, qfrag[ks], false, kf[nt], (short)0, S[nt], false, false);
    }

    // ---- scale + causal mask on diagonal block
    const bool diag = (kb == qb);
#pragma unroll
    for (int nt = 0; nt < 4; ++nt) {
      const int key = kv0 + nt * 16 + l15;
#pragma unroll
      for (int j = 0; j < 8; ++j) {
        float z = S[nt][j] * kScale;
        if (diag) {
          const int qi = q0 + wid * 16 + j + 8 * half;
          if (key > qi) z = -3.0e38f;
        }
        S[nt][j] = z;
      }
    }

    // ---- online softmax per row (row = j + 8*half lives on a 16-lane group)
#pragma unroll
    for (int j = 0; j < 8; ++j) {
      float m = fmaxf(fmaxf(S[0][j], S[1][j]), fmaxf(S[2][j], S[3][j]));
#pragma unroll
      for (int o = 8; o > 0; o >>= 1) m = fmaxf(m, __shfl_xor(m, o, 32));
      const float mnew  = fmaxf(rowmax[j], m);
      const float alpha = __expf(rowmax[j] - mnew);
      rowmax[j] = mnew;
      float ls = 0.0f;
#pragma unroll
      for (int nt = 0; nt < 4; ++nt) {
        const float p = __expf(S[nt][j] - mnew);
        S[nt][j] = p;
        ls += p;
      }
#pragma unroll
      for (int o = 8; o > 0; o >>= 1) ls += __shfl_xor(ls, o, 32);
      rowsum[j] = rowsum[j] * alpha + ls;
#pragma unroll
      for (int nt = 0; nt < 8; ++nt) Oacc[nt][j] *= alpha;
    }

    // ---- P (C-layout) -> per-wave LDS tile -> A-layout fragments
#pragma unroll
    for (int nt = 0; nt < 4; ++nt)
#pragma unroll
      for (int j = 0; j < 8; ++j)
        Plds[wid][j + 8 * half][nt * 16 + l15] = (__bf16)S[nt][j];

    // ---- O += P * V   (fragments grouped so WMMAs issue in bursts of 4)
#pragma unroll
    for (int ks = 0; ks < 2; ++ks) {
      v8u u;
#pragma unroll
      for (int p = 0; p < 8; ++p) {
        const int kk = ks * 32 + 2 * p + 16 * half;
        u[p] = *(const unsigned int*)&Plds[wid][l15][kk];
      }
      const v16bf pf = __builtin_bit_cast(v16bf, u);
#pragma unroll
      for (int g = 0; g < 2; ++g) {
        v16bf vf[4];
#pragma unroll
        for (int i = 0; i < 4; ++i) {
          const int nt = g * 4 + i;
#pragma unroll
          for (int e = 0; e < 16; ++e)
            vf[i][e] = Vs[cur][ks * 32 + e + 16 * half][nt * 16 + l15];
        }
#pragma unroll
        for (int i = 0; i < 4; ++i)
          Oacc[g * 4 + i] = __builtin_amdgcn_wmma_f32_16x16x32_bf16(
              false, pf, false, vf[i], (short)0, Oacc[g * 4 + i], false, false);
      }
    }
    __syncthreads();  // reads of [cur] done before next iter's DMA writes it
  }

  // ---- epilogue: normalize and write y[b][t][h*128 + d]
#pragma unroll
  for (int j = 0; j < 8; ++j) {
    const float inv = 1.0f / rowsum[j];
    const int t = q0 + wid * 16 + j + 8 * half;
    __bf16* yrow = y + ((size_t)(b * kT + t) * (kNH * kHD)) + h * kHD;
#pragma unroll
    for (int nt = 0; nt < 8; ++nt)
      yrow[nt * 16 + l15] = (__bf16)(Oacc[nt][j] * inv);
  }
}

// ---------------- host launch ----------------
extern "C" void kernel_launch(void* const* d_in, const int* in_sizes, int n_in,
                              void* d_out, int out_size, void* d_ws, size_t ws_size,
                              hipStream_t stream) {
  const float* x   = (const float*)d_in[0];
  const float* wq  = (const float*)d_in[1];
  const float* wk  = (const float*)d_in[2];
  const float* wv  = (const float*)d_in[3];
  const float* wo  = (const float*)d_in[4];
  const float* qnw = (const float*)d_in[5];
  const float* knw = (const float*)d_in[6];
  float* out = (float*)d_out;

  const int BT = kB * kT;  // 4096 rows

  char* ws = (char*)d_ws;
  size_t off = 0;
  auto alloc = [&](size_t bytes) {
    void* p = ws + off;
    off += (bytes + 255) & ~(size_t)255;
    return p;
  };
  __bf16* xb   = (__bf16*)alloc((size_t)BT * kD * 2);
  __bf16* wqb  = (__bf16*)alloc((size_t)kD * (kNH * kHD) * 2);
  __bf16* wkb  = (__bf16*)alloc((size_t)kD * (kNKV * kHD) * 2);
  __bf16* wvb  = (__bf16*)alloc((size_t)kD * (kNKV * kHD) * 2);
  __bf16* wob  = (__bf16*)alloc((size_t)(kNH * kHD) * kD * 2);
  __bf16* qraw = (__bf16*)alloc((size_t)BT * (kNH * kHD) * 2);
  __bf16* kraw = (__bf16*)alloc((size_t)BT * (kNKV * kHD) * 2);
  __bf16* vraw = (__bf16*)alloc((size_t)BT * (kNKV * kHD) * 2);
  __bf16* qarr = (__bf16*)alloc((size_t)kB * kNH * kT * kHD * 2);
  __bf16* karr = (__bf16*)alloc((size_t)kB * kNKV * kT * kHD * 2);
  __bf16* varr = (__bf16*)alloc((size_t)kB * kNKV * kT * kHD * 2);
  __bf16* ybuf = (__bf16*)alloc((size_t)BT * (kNH * kHD) * 2);

  // 1) bf16 conversions
  auto cvt = [&](const float* src, __bf16* dst, int n) {
    cvt_f32_bf16<<<(n + 255) / 256, 256, 0, stream>>>(src, dst, n);
  };
  cvt(x,  xb,  BT * kD);
  cvt(wq, wqb, kD * kNH * kHD);
  cvt(wk, wkb, kD * kNKV * kHD);
  cvt(wv, wvb, kD * kNKV * kHD);
  cvt(wo, wob, kNH * kHD * kD);

  // 2) QKV projections (bf16 WMMA, bf16 out)
  gemm_bf16_wmma<__bf16><<<dim3(BT / 128, (kNH * kHD) / 128), 256, 0, stream>>>(
      xb, wqb, qraw, BT, kNH * kHD, kD);
  gemm_bf16_wmma<__bf16><<<dim3(BT / 128, (kNKV * kHD) / 128), 256, 0, stream>>>(
      xb, wkb, kraw, BT, kNKV * kHD, kD);
  gemm_bf16_wmma<__bf16><<<dim3(BT / 128, (kNKV * kHD) / 128), 256, 0, stream>>>(
      xb, wvb, vraw, BT, kNKV * kHD, kD);

  // 3) RMSNorm + RoPE (q, k) and V layout permute
  norm_rope_kernel<<<kB * kNH * kT, 128, 0, stream>>>(qraw, qnw, qarr, kNH);
  norm_rope_kernel<<<kB * kNKV * kT, 128, 0, stream>>>(kraw, knw, karr, kNKV);
  vperm_kernel<<<(kB * kT * kNKV * kHD) / 256, 256, 0, stream>>>(vraw, varr);

  // 4) causal flash attention
  flash_attn_kernel<<<kB * kNH * (kT / 64), 128, 0, stream>>>(qarr, karr, varr, ybuf);

  // 5) output projection -> f32 d_out
  gemm_bf16_wmma<float><<<dim3(BT / 128, kD / 128), 256, 0, stream>>>(
      ybuf, wob, out, BT, kD, kNH * kHD);
}
